// BigramLanguageModel_87213605913026
// MI455X (gfx1250) — matrix operand order repaired and checked by reading, not compiled, not run on hardware
//
#include <hip/hip_runtime.h>
#include <hip/hip_bf16.h>

// ---------------- types ----------------
typedef _Float16 h16;
typedef __attribute__((ext_vector_type(16))) _Float16 v16h;
typedef __attribute__((ext_vector_type(8)))  float    v8f;

#define NTOK 32768   // B*T = 128*256
#define TSEQ 256
#define CEMB 384
#define NHEAD 6
#define HSZ 4
#define VOCAB 65

// ---- CDNA5 Tensor Data Mover availability (device pass only) ----
#if defined(__HIP_DEVICE_COMPILE__) && defined(__gfx1250__)
#if __has_builtin(__builtin_amdgcn_tensor_load_to_lds) && __has_builtin(__builtin_amdgcn_s_wait_tensorcnt)
#define USE_TDM 1
#else
#define USE_TDM 0
#endif
#else
#define USE_TDM 0
#endif

#if USE_TDM
typedef __attribute__((ext_vector_type(4))) unsigned int u32x4;
typedef __attribute__((ext_vector_type(4))) int          i32x4;
typedef __attribute__((ext_vector_type(8))) int          i32x8;

// LDS byte offset of a __shared__ object (addrspacecast generic->AS3, then ptrtoint)
__device__ inline unsigned lds_off_of(const void* p) {
    return (unsigned)(unsigned long long)(__attribute__((address_space(3))) const char*)p;
}

// Issue one 2D TDM tile load: tile (tile1 rows x tile0 elems) of 2-byte elements from
// row-major global (row stride = `stride` elems) into LDS at lds_off, with LDS padding
// of 4 DWORDs after every 16 DWORDs stored (i.e. 32 halves data + 8 halves pad = 40-half rows).
// tensor dims td0/td1 = remaining valid extent from the tile start; HW zero-fills OOB reads.
__device__ inline void tdm_load_2d(unsigned lds_off, const void* gptr,
                                   unsigned td0, unsigned td1,
                                   unsigned tile0, unsigned tile1, unsigned stride) {
    unsigned long long ga = (unsigned long long)gptr;
    u32x4 g0;
    g0.x = 1u;                                                  // count=1, user descriptor
    g0.y = lds_off;                                             // lds_addr
    g0.z = (unsigned)ga;                                        // global_addr[31:0]
    g0.w = (unsigned)((ga >> 32) & 0x01FFFFFFu) | 0x80000000u;  // addr[56:32] | type=2
    i32x8 g1;
    g1[0] = (int)((1u << 16)      // data_size = 2 bytes
                | (1u << 20)      // pad_enable
                | (3u << 22)      // pad_interval: 16 DWORDs
                | (3u << 25));    // pad_amount: 4 DWORDs
    g1[1] = (int)((td0 & 0xFFFFu) << 16);                       // tensor_dim0[15:0] @bits63:48
    g1[2] = (int)((td0 >> 16) | ((td1 & 0xFFFFu) << 16));       // td0 hi | td1 lo
    g1[3] = (int)((td1 >> 16) | (tile0 << 16));                 // td1 hi | tile_dim0
    g1[4] = (int)(tile1 & 0xFFFFu);                             // tile_dim1 | tile_dim2=0
    g1[5] = (int)stride;                                        // tensor_dim0_stride lo32
    g1[6] = 0;                                                  // stride hi | dim1_stride lo
    g1[7] = 0;
    i32x4 z4 = {0, 0, 0, 0};
#if __clang_major__ >= 23
    i32x8 z8 = {0, 0, 0, 0, 0, 0, 0, 0};
    __builtin_amdgcn_tensor_load_to_lds(g0, g1, z4, z4, z8, 0);
#else
    __builtin_amdgcn_tensor_load_to_lds(g0, g1, z4, z4, 0);
#endif
}
#endif // USE_TDM

// ---------------- embedding: x = tok_emb[idx] + pos_emb, store f16 ----------------
__global__ void embed_k(const int* __restrict__ idx, const float* __restrict__ tok,
                        const float* __restrict__ pos, h16* __restrict__ xh) {
    int i = blockIdx.x * blockDim.x + threadIdx.x;   // exact: NTOK*CEMB threads
    int t = i / CEMB;
    int c = i - t * CEMB;
    int tt = t % TSEQ;
    float v = tok[idx[t] * CEMB + c] + pos[tt * CEMB + c];
    xh[i] = (h16)v;
}

// ---- pack Wq/Wk/Wv [6,384,4] -> TRANSPOSED [72,384] f16 (rows: q0..23,k24..47,v48..71) ----
__global__ void pack_qkvT_k(const float* __restrict__ Wq, const float* __restrict__ Wk,
                            const float* __restrict__ Wv, h16* __restrict__ wqkvT) {
    int i = blockIdx.x * blockDim.x + threadIdx.x;   // 72*384 exact
    int j = i / CEMB;
    int c = i - j * CEMB;
    int g = j / 24;
    int jj = j - g * 24;
    int h = jj >> 2, d = jj & 3;
    const float* W = (g == 0) ? Wq : ((g == 1) ? Wk : Wv);
    wqkvT[i] = (h16)W[(h * CEMB + c) * HSZ + d];
}

// ---------------- f32 [K,N] -> f16 transposed [N,K] ----------------
__global__ void cvtT_f32_f16_k(const float* __restrict__ s, h16* __restrict__ d,
                               int K, int N) {
    int i = blockIdx.x * blockDim.x + threadIdx.x;
    if (i < K * N) {
        int k = i / N;
        int n = i - k * N;
        d[(size_t)n * K + k] = (h16)s[i];
    }
}

// ---------------- tiled WMMA GEMM: C[M,N] = A[M,K] @ Bt[N,K]^T (+bias)(+relu) ----------------
// A f16 [M,K] row-major; Bt f16 [N,K] row-major (pre-transposed weight).
// Block tile 128x64, 256 threads = 8 waves (4M x 2N); each wave: 32x32 slab = 4 WMMA accums.
// TDM path: wave 0 DMAs both tiles for step s+1 (ping-pong LDS) while all waves compute
// step s; one barrier + s_wait_tensorcnt per k-step; HW OOB zero-fill handles K/N tails.
#define TM 128
#define TN 64
#define TK 32
#define ASTR 40
#define BSTR 40

template <bool KA, bool NA>
__global__ __launch_bounds__(256) void gemm_t(
    const h16* __restrict__ A, const h16* __restrict__ Bt,
    const float* __restrict__ bias, float* __restrict__ Cf, h16* __restrict__ Ch,
    int M, int N, int K, int relu)
{
    const int tid  = threadIdx.x;
    const int lane = tid & 31;
    const int wave = tid >> 5;
    const int wm   = wave & 3;        // 4 row blocks of 32
    const int wn   = wave >> 2;       // 2 col blocks of 32
    const int hi   = lane >> 4;
    const int ln   = lane & 15;
    const int m0   = blockIdx.y * TM;
    const int n0   = blockIdx.x * TN;

    v8f c00 = {}, c01 = {}, c10 = {}, c11 = {};

#if USE_TDM
    __shared__ __attribute__((aligned(16))) h16 As[2][TM * ASTR];
    __shared__ __attribute__((aligned(16))) h16 Bs[2][TN * BSTR];

    const int nsteps = (K + TK - 1) / TK;
    const h16* gA = A  + (size_t)m0 * K;   // tile start, k advances by TK per step
    const h16* gB = Bt + (size_t)n0 * K;

    if (wave == 0) {
        tdm_load_2d(lds_off_of(&As[0][0]), gA, (unsigned)K, (unsigned)(M - m0), TK, TM, (unsigned)K);
        tdm_load_2d(lds_off_of(&Bs[0][0]), gB, (unsigned)K, (unsigned)(N - n0), TK, TN, (unsigned)K);
    }
    for (int s = 0; s < nsteps; ++s) {
        const int cur = s & 1;
        if (wave == 0) __builtin_amdgcn_s_wait_tensorcnt(0);   // tile s resident
        __syncthreads();                                       // all waves: data ready; buf cur^1 free
        if (wave == 0 && (s + 1) < nsteps) {
            int k1 = (s + 1) * TK;
            tdm_load_2d(lds_off_of(&As[cur ^ 1][0]), gA + k1,
                        (unsigned)(K - k1), (unsigned)(M - m0), TK, TM, (unsigned)K);
            tdm_load_2d(lds_off_of(&Bs[cur ^ 1][0]), gB + k1,
                        (unsigned)(K - k1), (unsigned)(N - n0), TK, TN, (unsigned)K);
        }
        const h16* Ab = &As[cur][0];
        const h16* Bb = &Bs[cur][0];
#else
    __shared__ __attribute__((aligned(16))) h16 As[TM * ASTR];
    __shared__ __attribute__((aligned(16))) h16 Bs[TN * BSTR];
    // A staging: row = tid>>1 (0..127), 16 contiguous k at (tid&1)*16
    const int ra = tid >> 1;
    const int ka = (tid & 1) << 4;
    // B staging: row n = tid>>2 (0..63), 8 contiguous k at (tid&3)*8
    const int rb = tid >> 2;
    const int kb = (tid & 3) << 3;

    for (int k0 = 0; k0 < K; k0 += TK) {
        __syncthreads();
        {
            const h16* src = A + (size_t)(m0 + ra) * K + (size_t)(k0 + ka);
            h16* dst = &As[ra * ASTR + ka];
            if (KA || (k0 + ka + 16 <= K)) {
                *reinterpret_cast<uint4*>(dst)     = *reinterpret_cast<const uint4*>(src);
                *reinterpret_cast<uint4*>(dst + 8) = *reinterpret_cast<const uint4*>(src + 8);
            } else {
#pragma unroll
                for (int j = 0; j < 16; ++j)
                    dst[j] = (k0 + ka + j < K) ? src[j] : (h16)0.f;
            }
        }
        {
            h16* dst = &Bs[rb * BSTR + kb];
            if (NA || (n0 + rb < N)) {
                const h16* src = Bt + (size_t)(n0 + rb) * K + (size_t)(k0 + kb);
                if (KA || (k0 + kb + 8 <= K)) {
                    *reinterpret_cast<uint4*>(dst) = *reinterpret_cast<const uint4*>(src);
                } else {
#pragma unroll
                    for (int j = 0; j < 8; ++j)
                        dst[j] = (k0 + kb + j < K) ? src[j] : (h16)0.f;
                }
            } else {
                uint4 z = {0u, 0u, 0u, 0u};
                *reinterpret_cast<uint4*>(dst) = z;
            }
        }
        __syncthreads();
        const h16* Ab = &As[0];
        const h16* Bb = &Bs[0];
#endif
        // ---- fragments from LDS (k-pairs contiguous -> ds_load_b128) ----
        union F { unsigned int u[8]; v16h v; } a0, a1, b0, b1;
#pragma unroll
        for (int i = 0; i < 8; ++i) {
            // A 16x32 f16 layout: m = ln; k = (i/4)*16 + hi*8 + (i%4)*2 (+h)
            int kA = ((i >> 2) << 4) | (hi << 3) | ((i & 3) << 1);
            a0.u[i] = *reinterpret_cast<const unsigned int*>(&Ab[(wm * 32 + ln) * ASTR + kA]);
            a1.u[i] = *reinterpret_cast<const unsigned int*>(&Ab[(wm * 32 + 16 + ln) * ASTR + kA]);
            // B 32x16 f16 layout: n = ln; k = (i/4)*16 + hi*8 + (i%4)*2 (+h)
            int kB = ((i >> 2) << 4) | (hi << 3) | ((i & 3) << 1);
            b0.u[i] = *reinterpret_cast<const unsigned int*>(&Bb[(wn * 32 + ln) * BSTR + kB]);
            b1.u[i] = *reinterpret_cast<const unsigned int*>(&Bb[(wn * 32 + 16 + ln) * BSTR + kB]);
        }
        c00 = __builtin_amdgcn_wmma_f32_16x16x32_f16(false, a0.v, false, b0.v, (short)0, c00, false, false);
        c01 = __builtin_amdgcn_wmma_f32_16x16x32_f16(false, a0.v, false, b1.v, (short)0, c01, false, false);
        c10 = __builtin_amdgcn_wmma_f32_16x16x32_f16(false, a1.v, false, b0.v, (short)0, c10, false, false);
        c11 = __builtin_amdgcn_wmma_f32_16x16x32_f16(false, a1.v, false, b1.v, (short)0, c11, false, false);
    }

    // ---- epilogue ----
    v8f acc[2][2] = {{c00, c01}, {c10, c11}};
#pragma unroll
    for (int sub = 0; sub < 2; ++sub) {
#pragma unroll
        for (int cb = 0; cb < 2; ++cb) {
            int n = n0 + wn * 32 + cb * 16 + ln;
            if (NA || n < N) {
                float bv = bias ? bias[n] : 0.0f;
#pragma unroll
                for (int r = 0; r < 8; ++r) {
                    int m = m0 + wm * 32 + sub * 16 + (hi << 3) + r;
                    float v = acc[sub][cb][r] + bv;
                    if (relu) v = v > 0.f ? v : 0.f;
                    if (Cf) Cf[(size_t)m * N + n] = v;
                    if (Ch) Ch[(size_t)m * N + n] = (h16)v;
                }
            }
        }
    }
}

// ---------------- causal attention, head_size 4, online softmax ----------------
__global__ void attn_k(const float* __restrict__ qkv, h16* __restrict__ attn_h) {
    const int bh = blockIdx.x;
    const int b  = bh / NHEAD;
    const int h  = bh - b * NHEAD;
    const int t  = threadIdx.x;

    __shared__ float ks[TSEQ][HSZ];
    __shared__ float vs[TSEQ][HSZ];

    const float* base = qkv + (size_t)b * TSEQ * 72;
#pragma unroll
    for (int d = 0; d < HSZ; ++d) {
        ks[t][d] = base[t * 72 + 24 + h * HSZ + d];
        vs[t][d] = base[t * 72 + 48 + h * HSZ + d];
    }
    __syncthreads();

    float q0 = base[t * 72 + h * HSZ + 0];
    float q1 = base[t * 72 + h * HSZ + 1];
    float q2 = base[t * 72 + h * HSZ + 2];
    float q3 = base[t * 72 + h * HSZ + 3];
    const float scale = 0.051031036307982884f;   // 384^-0.5 (reference scales by C)

    float mrun = -3.0e38f, lrun = 0.f;
    float acc0 = 0.f, acc1 = 0.f, acc2 = 0.f, acc3 = 0.f;
    for (int s = 0; s <= t; ++s) {
        float sc = (q0 * ks[s][0] + q1 * ks[s][1] + q2 * ks[s][2] + q3 * ks[s][3]) * scale;
        float mn = fmaxf(mrun, sc);
        float corr = __expf(mrun - mn);
        float e    = __expf(sc - mn);
        lrun = lrun * corr + e;
        acc0 = acc0 * corr + e * vs[s][0];
        acc1 = acc1 * corr + e * vs[s][1];
        acc2 = acc2 * corr + e * vs[s][2];
        acc3 = acc3 * corr + e * vs[s][3];
        mrun = mn;
    }
    float inv = 1.f / lrun;
    size_t o = ((size_t)b * TSEQ + t) * (NHEAD * HSZ) + h * HSZ;
    attn_h[o + 0] = (h16)(acc0 * inv);
    attn_h[o + 1] = (h16)(acc1 * inv);
    attn_h[o + 2] = (h16)(acc2 * inv);
    attn_h[o + 3] = (h16)(acc3 * inv);
}

// ---------------- per-row cross entropy ----------------
__global__ void loss_rows_k(const float* __restrict__ logits, const int* __restrict__ targets,
                            float* __restrict__ rowloss) {
    int i = blockIdx.x * blockDim.x + threadIdx.x;
    const float* row = logits + (size_t)i * VOCAB;
    float mx = row[0];
#pragma unroll 4
    for (int j = 1; j < VOCAB; ++j) mx = fmaxf(mx, row[j]);
    float s = 0.f;
#pragma unroll 4
    for (int j = 0; j < VOCAB; ++j) s += __expf(row[j] - mx);
    int tgt = targets[i];
    rowloss[i] = -(row[tgt] - mx - __logf(s));
}

// ---------------- deterministic fixed-order mean reduce ----------------
__global__ void loss_reduce_k(const float* __restrict__ rowloss, float* __restrict__ out) {
    __shared__ float sm[256];
    float s = 0.f;
    for (int j = threadIdx.x; j < NTOK; j += 256) s += rowloss[j];
    sm[threadIdx.x] = s;
    __syncthreads();
    for (int off = 128; off > 0; off >>= 1) {
        if ((int)threadIdx.x < off) sm[threadIdx.x] += sm[threadIdx.x + off];
        __syncthreads();
    }
    if (threadIdx.x == 0) out[0] = sm[0] / (float)NTOK;
}

// ---------------- launch ----------------
extern "C" void kernel_launch(void* const* d_in, const int* in_sizes, int n_in,
                              void* d_out, int out_size, void* d_ws, size_t ws_size,
                              hipStream_t stream) {
    (void)in_sizes; (void)n_in; (void)out_size; (void)ws_size;
    const int*   idx     = (const int*)  d_in[0];
    const int*   targets = (const int*)  d_in[1];
    const float* tok     = (const float*)d_in[2];
    const float* pos     = (const float*)d_in[3];
    const float* Wq      = (const float*)d_in[4];
    const float* Wk      = (const float*)d_in[5];
    const float* Wv      = (const float*)d_in[6];
    const float* Wproj   = (const float*)d_in[7];
    const float* bproj   = (const float*)d_in[8];
    const float* W1      = (const float*)d_in[9];
    const float* b1      = (const float*)d_in[10];
    const float* W2      = (const float*)d_in[11];
    const float* b2      = (const float*)d_in[12];
    const float* Wlm     = (const float*)d_in[13];
    const float* blm     = (const float*)d_in[14];

    char* wsp = (char*)d_ws;
    auto alloc = [&](size_t bytes) -> void* {
        void* p = (void*)wsp;
        wsp += (bytes + 255) & ~(size_t)255;
        return p;
    };
    h16*   x_h     = (h16*)  alloc((size_t)NTOK * CEMB * 2);
    h16*   wqkvT   = (h16*)  alloc((size_t)72 * CEMB * 2);
    float* qkv     = (float*)alloc((size_t)NTOK * 72 * 4);
    h16*   attn_h  = (h16*)  alloc((size_t)NTOK * 24 * 2);
    h16*   wprojT  = (h16*)  alloc((size_t)CEMB * 24 * 2);
    h16*   x2_h    = (h16*)  alloc((size_t)NTOK * CEMB * 2);
    h16*   w1T     = (h16*)  alloc((size_t)1536 * CEMB * 2);
    h16*   h1_h    = (h16*)  alloc((size_t)8192 * 1536 * 2);
    h16*   w2T     = (h16*)  alloc((size_t)CEMB * 1536 * 2);
    h16*   x3_h    = (h16*)  alloc((size_t)NTOK * CEMB * 2);
    h16*   wlmT    = (h16*)  alloc((size_t)VOCAB * CEMB * 2);
    float* rowloss = (float*)alloc((size_t)NTOK * 4);

    float* logits = (float*)d_out;
    float* lossp  = logits + (size_t)NTOK * VOCAB;

    embed_k    <<<(NTOK * CEMB) / 256, 256, 0, stream>>>(idx, tok, pos, x_h);
    pack_qkvT_k<<<(72 * CEMB) / 256,   256, 0, stream>>>(Wq, Wk, Wv, wqkvT);
    cvtT_f32_f16_k<<<(24 * CEMB + 255) / 256,    256, 0, stream>>>(Wproj, wprojT, 24, CEMB);
    cvtT_f32_f16_k<<<(CEMB * 1536 + 255) / 256,  256, 0, stream>>>(W1, w1T, CEMB, 1536);
    cvtT_f32_f16_k<<<(1536 * CEMB + 255) / 256,  256, 0, stream>>>(W2, w2T, 1536, CEMB);
    cvtT_f32_f16_k<<<(CEMB * VOCAB + 255) / 256, 256, 0, stream>>>(Wlm, wlmT, CEMB, VOCAB);

    // QKV: [NTOK,384] @ [384,72] -> f32
    gemm_t<true, false><<<dim3(2, NTOK / TM), 256, 0, stream>>>(
        x_h, wqkvT, nullptr, qkv, nullptr, NTOK, 72, CEMB, 0);
    attn_k<<<128 * NHEAD, 256, 0, stream>>>(qkv, attn_h);
    // proj: [NTOK,24] @ [24,384] + bproj -> f16
    gemm_t<false, true><<<dim3(6, NTOK / TM), 256, 0, stream>>>(
        attn_h, wprojT, bproj, nullptr, x2_h, NTOK, CEMB, 24, 0);
    // FFN, chunked over M (hot, fully aligned)
    for (int c = 0; c < 4; ++c) {
        const h16* a = x2_h + (size_t)c * 8192 * CEMB;
        h16*       o = x3_h + (size_t)c * 8192 * CEMB;
        gemm_t<true, true><<<dim3(24, 8192 / TM), 256, 0, stream>>>(
            a, w1T, b1, nullptr, h1_h, 8192, 1536, CEMB, 1);
        gemm_t<true, true><<<dim3(6, 8192 / TM), 256, 0, stream>>>(
            h1_h, w2T, b2, nullptr, o, 8192, CEMB, 1536, 0);
    }
    // LM head -> logits
    gemm_t<true, false><<<dim3(2, NTOK / TM), 256, 0, stream>>>(
        x3_h, wlmT, blm, logits, nullptr, NTOK, VOCAB, CEMB, 0);
    loss_rows_k  <<<NTOK / 256, 256, 0, stream>>>(logits, targets, rowloss);
    loss_reduce_k<<<1, 256, 0, stream>>>(rowloss, lossp);
}
